// SRGAN_24283745091608
// MI455X (gfx1250) — matrix operand order repaired and checked by reading, not compiled, not run on hardware
//
#include <hip/hip_runtime.h>
#include <hip/hip_bf16.h>

// Problem constants (match reference)
#define BATCH 32
#define HH 64
#define WW 64
#define CC 64

typedef __attribute__((ext_vector_type(16))) _Float16 v16h;
typedef __attribute__((ext_vector_type(8)))  _Float16 v8h;
typedef __attribute__((ext_vector_type(8)))  float    v8f;

static __device__ __forceinline__ v16h cat8(v8h lo, v8h hi) {
  return __builtin_shufflevector(lo, hi, 0,1,2,3,4,5,6,7,8,9,10,11,12,13,14,15);
}

static __device__ __forceinline__ v8h cvt8(float4 a, float4 b) {
  v8h r;
  r[0]=(_Float16)a.x; r[1]=(_Float16)a.y; r[2]=(_Float16)a.z; r[3]=(_Float16)a.w;
  r[4]=(_Float16)b.x; r[5]=(_Float16)b.y; r[6]=(_Float16)b.z; r[7]=(_Float16)b.w;
  return r;
}

// ---------------------------------------------------------------------------
// Pack 3x3 64->64 f32 weights (O,I,3,3) into WMMA f16 B-fragment layout.
// K ordering: K = tap*64 + ci  (tap = ky*3+kx), 18 K-blocks of 32.
// Per (kb, nb): 32 lanes x 16 halves contiguous; lane<16 -> K_local 0..7,16..23,
// lane>=16 -> K_local 8..15,24..31 (matches V_WMMA B 32x16 layout, 8 VGPRs).
// ---------------------------------------------------------------------------
__global__ void pack_w3(const float* __restrict__ w, _Float16* __restrict__ wp) {
  int i = blockIdx.x * 256 + threadIdx.x;
  if (i >= 18 * 4 * 32 * 16) return;
  int q    = i & 15;
  int lane = (i >> 4) & 31;
  int nb   = (i >> 9) & 3;
  int kb   = i >> 11;
  int K = kb * 32 + ((lane >= 16) ? 8 : 0) + ((q < 8) ? q : q + 8);
  int N = nb * 16 + (lane & 15);
  int tap = K >> 6, ci = K & 63;
  int ky = tap / 3, kx = tap % 3;
  wp[i] = (_Float16)w[((N * 64 + ci) * 3 + ky) * 3 + kx];
}

// Pack 1x1 256->64 weights with per-batch channel-attention scale folded in.
// Layout: [b][kb(8)][nb(4)][lane(32)][q(16)]
__global__ void pack_w1(const float* __restrict__ w, const float* __restrict__ scale,
                        _Float16* __restrict__ wp) {
  int i = blockIdx.x * 256 + threadIdx.x;  // 2048*256 = 524288 exact
  int q    = i & 15;
  int lane = (i >> 4) & 31;
  int nb   = (i >> 9) & 3;
  int kb   = (i >> 11) & 7;
  int b    = i >> 14;
  int K = kb * 32 + ((lane >= 16) ? 8 : 0) + ((q < 8) ? q : q + 8);
  int N = nb * 16 + (lane & 15);
  wp[i] = (_Float16)(w[N * 256 + K] * scale[b * 256 + K]);
}

// ---------------------------------------------------------------------------
// conv1: 1->64 3x3 + bias + relu (tiny, scalar). in NCHW C=1, out NHWC f32.
// ---------------------------------------------------------------------------
__global__ void conv1_k(const float* __restrict__ x, const float* __restrict__ w,
                        const float* __restrict__ b, float* __restrict__ out) {
  int i = blockIdx.x * 256 + threadIdx.x;  // pixel id, 512*256 = 131072 exact
  if (i >= BATCH * HH * WW) return;
  int wI = i & 63, hI = (i >> 6) & 63, bI = i >> 12;
  float patch[9];
#pragma unroll
  for (int ky = 0; ky < 3; ++ky)
#pragma unroll
    for (int kx = 0; kx < 3; ++kx) {
      int h2 = hI + ky - 1, w2 = wI + kx - 1;
      patch[ky * 3 + kx] = (h2 >= 0 && h2 < HH && w2 >= 0 && w2 < WW)
                               ? x[(bI << 12) + (h2 << 6) + w2] : 0.f;
    }
  float* op = out + (size_t)i * 64;
#pragma unroll 4
  for (int o = 0; o < 64; ++o) {
    float a = b[o];
#pragma unroll
    for (int t = 0; t < 9; ++t) a += patch[t] * w[o * 9 + t];
    op[o] = fmaxf(a, 0.f);
  }
}

// ---------------------------------------------------------------------------
// Main WMMA implicit-GEMM 3x3 conv, 64->64, NHWC f32 in/out.
// Workgroup: 256 thr = 8 waves; tile 8 rows x 16 cols x 64 out-ch.
// Wave j computes pixel row h0+j: M = column (0..15), N = out channel.
// LDS: halo tile 10x18x64 in f16 (stride 72 halves to spread banks).
// ---------------------------------------------------------------------------
__global__ __launch_bounds__(256) void conv3_wmma(
    const float* __restrict__ in, const _Float16* __restrict__ wp,
    const float* __restrict__ bias, const float* __restrict__ res,
    float* __restrict__ out, int doRelu, int doAdd) {
  __shared__ __align__(16) _Float16 sIn[10][18][72];
  int tid = threadIdx.x;
  int lane = tid & 31, wv = tid >> 5;
  int wb0 = blockIdx.x * 16, hb0 = blockIdx.y * 8, bI = blockIdx.z;

  // Stage input halo tile (all 64 channels) as f16.
  for (int idx = tid; idx < 180 * 4; idx += 256) {
    int chunk = idx & 3, pix = idx >> 2;
    int pr = pix / 18, pc = pix - pr * 18;
    int gh = hb0 + pr - 1, gw = wb0 + pc - 1;
    _Float16* dst = &sIn[pr][pc][chunk * 16];
    if (gh >= 0 && gh < HH && gw >= 0 && gw < WW) {
      const float4* sp =
          (const float4*)(in + (((size_t)(bI * HH + gh)) * WW + gw) * 64 + chunk * 16);
      float4 A0 = sp[0], A1 = sp[1], A2 = sp[2], A3 = sp[3];
      *(v8h*)dst       = cvt8(A0, A1);
      *(v8h*)(dst + 8) = cvt8(A2, A3);
    } else {
      v8h z = {};
      *(v8h*)dst = z; *(v8h*)(dst + 8) = z;
    }
  }
  __syncthreads();

  int m16 = lane & 15;
  int hiSel = (lane >= 16) ? 8 : 0;  // A: K-half select; C: M-row +8 select
  v8f z8 = {0.f,0.f,0.f,0.f,0.f,0.f,0.f,0.f};
  v8f acc[4] = {z8, z8, z8, z8};

  for (int kb = 0; kb < 18; ++kb) {
    int tap = kb >> 1, cbase = (kb & 1) * 32;
    int dy = tap / 3, dx = tap % 3;  // halo row = wv+dy (the -1 pad cancels +1 origin)
    const _Float16* ap = &sIn[wv + dy][m16 + dx][cbase + hiSel];
    v8h alo = *(const v8h*)ap;        // K_local (hiSel?8:0)+0..7
    v8h ahi = *(const v8h*)(ap + 16); // K_local +16
    v16h a = cat8(alo, ahi);
#pragma unroll
    for (int nb = 0; nb < 4; ++nb) {
      const _Float16* bp = wp + ((size_t)(kb * 4 + nb) * 32 + lane) * 16;
      v16h bm = cat8(*(const v8h*)bp, *(const v8h*)(bp + 8));
      acc[nb] = __builtin_amdgcn_wmma_f32_16x16x32_f16(
          false, a, false, bm, (short)0, acc[nb], false, false);
    }
  }

  int h = hb0 + wv;
#pragma unroll
  for (int nb = 0; nb < 4; ++nb) {
    int co = nb * 16 + m16;
    float bv = bias[co];
#pragma unroll
    for (int r = 0; r < 8; ++r) {
      int m = r + hiSel;  // C layout: lanes 0-15 M=r, lanes 16-31 M=r+8
      size_t oidx = (((size_t)(bI * HH + h)) * WW + (wb0 + m)) * 64 + co;
      float v = acc[nb][r] + bv;
      if (doRelu) v = fmaxf(v, 0.f);
      if (doAdd)  v += res[oidx];
      out[oidx] = v;
    }
  }
}

// ---------------------------------------------------------------------------
// 1x1 conv 256->64 via WMMA over 4 concatenated 64-ch sources (channel
// attention scale pre-folded into per-batch packed weights).
// out = relu(conv) + res  (conv7 / conv13 pattern).
// ---------------------------------------------------------------------------
__global__ __launch_bounds__(256) void conv1x1_wmma(
    const float* __restrict__ s0, const float* __restrict__ s1,
    const float* __restrict__ s2, const float* __restrict__ s3,
    const _Float16* __restrict__ wp, const float* __restrict__ bias,
    const float* __restrict__ res, float* __restrict__ out) {
  int tid = threadIdx.x;
  int lane = tid & 31, wv = tid >> 5;
  int wb0 = blockIdx.x * 16, hb0 = blockIdx.y * 8, bI = blockIdx.z;
  int m16 = lane & 15;
  int hiSel = (lane >= 16) ? 8 : 0;
  size_t pixBase = (((size_t)(bI * HH + hb0 + wv)) * WW + (wb0 + m16)) * 64;
  const _Float16* wpb = wp + (size_t)bI * 16384;
  const float* srcs[4] = {s0, s1, s2, s3};

  v8f z8 = {0.f,0.f,0.f,0.f,0.f,0.f,0.f,0.f};
  v8f acc[4] = {z8, z8, z8, z8};

#pragma unroll
  for (int kb = 0; kb < 8; ++kb) {
    const float* src = srcs[kb >> 1];
    int cbase = (kb & 1) * 32;
    const float* p = src + pixBase + cbase + hiSel;
    const float4* f = (const float4*)p;
    float4 A0 = f[0], A1 = f[1];
    const float4* g = (const float4*)(p + 16);
    float4 A2 = g[0], A3 = g[1];
    v16h a = cat8(cvt8(A0, A1), cvt8(A2, A3));
#pragma unroll
    for (int nb = 0; nb < 4; ++nb) {
      const _Float16* bp = wpb + ((size_t)(kb * 4 + nb) * 32 + lane) * 16;
      v16h bm = cat8(*(const v8h*)bp, *(const v8h*)(bp + 8));
      acc[nb] = __builtin_amdgcn_wmma_f32_16x16x32_f16(
          false, a, false, bm, (short)0, acc[nb], false, false);
    }
  }

  int h = hb0 + wv;
#pragma unroll
  for (int nb = 0; nb < 4; ++nb) {
    int co = nb * 16 + m16;
    float bv = bias[co];
#pragma unroll
    for (int r = 0; r < 8; ++r) {
      int m = r + hiSel;
      size_t oidx = (((size_t)(bI * HH + h)) * WW + (wb0 + m)) * 64 + co;
      float v = fmaxf(acc[nb][r] + bv, 0.f) + res[oidx];
      out[oidx] = v;
    }
  }
}

// ---------------------------------------------------------------------------
// Channel attention: per-(b, concat-channel) spatial mean over 64x64.
// grid (32, 4): blockIdx.y picks source buffer (concat group).
// ---------------------------------------------------------------------------
__global__ void ca_mean(const float* __restrict__ s0, const float* __restrict__ s1,
                        const float* __restrict__ s2, const float* __restrict__ s3,
                        float* __restrict__ means) {
  __shared__ float red[256];
  int b = blockIdx.x, g = blockIdx.y;
  const float* src = (g == 0) ? s0 : (g == 1) ? s1 : (g == 2) ? s2 : s3;
  int tid = threadIdx.x;
  int c = tid & 63, part = tid >> 6;
  float sum = 0.f;
  for (int p = part; p < 4096; p += 4)
    sum += src[(((size_t)b * 4096) + p) * 64 + c];
  red[tid] = sum;
  __syncthreads();
  if (part == 0)
    means[b * 256 + g * 64 + c] =
        (red[c] + red[64 + c] + red[128 + c] + red[192 + c]) * (1.f / 4096.f);
}

// 256->16->256 squeeze-excite + sigmoid. One block per batch.
__global__ void ca_scale(const float* __restrict__ means, const float* __restrict__ w1,
                         const float* __restrict__ b1, const float* __restrict__ w2,
                         const float* __restrict__ b2, float* __restrict__ scale) {
  __shared__ float m[256];
  __shared__ float a1[16];
  int b = blockIdx.x, tid = threadIdx.x;
  m[tid] = means[b * 256 + tid];
  __syncthreads();
  if (tid < 16) {
    float s = b1[tid];
    for (int j = 0; j < 256; ++j) s += w1[tid * 256 + j] * m[j];
    a1[tid] = s;
  }
  __syncthreads();
  float t = b2[tid];
#pragma unroll
  for (int j = 0; j < 16; ++j) t += w2[tid * 16 + j] * a1[j];
  scale[b * 256 + tid] = 1.f / (1.f + expf(-t));
}

// ---------------------------------------------------------------------------
// Offset conv 64->18 3x3 + BN + tanh (scalar; ~1.4 GFLOP total).
// ---------------------------------------------------------------------------
__global__ void off_conv(const float* __restrict__ feat, const float* __restrict__ w,
                         const float* __restrict__ bias, const float* __restrict__ g,
                         const float* __restrict__ bb, const float* __restrict__ mm,
                         const float* __restrict__ vv, float* __restrict__ off) {
  int i = blockIdx.x * 256 + threadIdx.x;  // 9216*256 = 2359296 exact
  if (i >= BATCH * HH * WW * 18) return;
  int oc = i % 18;
  int pix = i / 18;
  int wI = pix & 63, hI = (pix >> 6) & 63, bI = pix >> 12;
  float acc = bias[oc];
  for (int ky = 0; ky < 3; ++ky) {
    int h2 = hI + ky - 1; if (h2 < 0 || h2 >= HH) continue;
    for (int kx = 0; kx < 3; ++kx) {
      int w2 = wI + kx - 1; if (w2 < 0 || w2 >= WW) continue;
      const float* fp = feat + (((size_t)(bI * HH + h2)) * WW + w2) * 64;
      const float* wq = w + oc * 576 + ky * 3 + kx;  // (O,I,3,3): + ci*9
#pragma unroll 8
      for (int ci = 0; ci < 64; ++ci) acc += fp[ci] * wq[ci * 9];
    }
  }
  float y = (acc - mm[oc]) * rsqrtf(vv[oc] + 1e-5f) * g[oc] + bb[oc];
  off[(size_t)pix * 18 + oc] = tanhf(y);
}

// ---------------------------------------------------------------------------
// Coordinate map + bilinear gather -> out (B, 192, 192, 64) f32.
// Thread handles 4 channels (float4) of one (pixel, tap) sample.
// ---------------------------------------------------------------------------
__global__ void deform_k(const float* __restrict__ feat, const float* __restrict__ off,
                         float* __restrict__ out) {
  size_t i = (size_t)blockIdx.x * 256 + threadIdx.x;  // 73728*256 = B*H*W*9*16
  if (i >= (size_t)BATCH * HH * WW * 9 * 16) return;
  int cq = (int)(i & 15);
  int p  = (int)((i >> 4) % 9);
  size_t pix = (i >> 4) / 9;
  int wI = (int)(pix & 63), hI = (int)((pix >> 6) & 63), bI = (int)(pix >> 12);
  float xo = off[pix * 18 + p];
  float yo = off[pix * 18 + 9 + p];
  int pk = p / 3, pw = p - pk * 3;
  // Reference quirk: gx uses p//3, gy uses p%3 (meshgrid 'ij' then flat).
  float x = (float)wI + (float)(pk - 1) + 2.f * xo;
  float y = (float)hI + (float)(pw - 1) + 2.f * yo;
  float xf = floorf(x), yf = floorf(y);
  int x0 = min(max((int)xf, 0), WW - 1), x1 = min(max((int)xf + 1, 0), WW - 1);
  int y0 = min(max((int)yf, 0), HH - 1), y1 = min(max((int)yf + 1, 0), HH - 1);
  float x0f = (float)x0, x1f = (float)x1, y0f = (float)y0, y1f = (float)y1;
  float wa = (x1f - x) * (y1f - y);
  float wb = (x1f - x) * (y - y0f);
  float wc = (x - x0f) * (y1f - y);
  float wd = (x - x0f) * (y - y0f);
  size_t base = (size_t)bI * HH * WW * 64 + (size_t)cq * 4;
  float4 va = *(const float4*)&feat[base + ((size_t)(y0 * WW + x0)) * 64];
  float4 vb = *(const float4*)&feat[base + ((size_t)(y1 * WW + x0)) * 64];
  float4 vc = *(const float4*)&feat[base + ((size_t)(y0 * WW + x1)) * 64];
  float4 vd = *(const float4*)&feat[base + ((size_t)(y1 * WW + x1)) * 64];
  float4 r;
  r.x = va.x * wa + vb.x * wb + vc.x * wc + vd.x * wd;
  r.y = va.y * wa + vb.y * wb + vc.y * wc + vd.y * wd;
  r.z = va.z * wa + vb.z * wb + vc.z * wc + vd.z * wd;
  r.w = va.w * wa + vb.w * wb + vc.w * wc + vd.w * wd;
  int row = hI * 3 + pk, col = wI * 3 + pw;
  *(float4*)&out[(((size_t)bI * 192 + row) * 192 + col) * 64 + cq * 4] = r;
}

// ---------------------------------------------------------------------------
// d_in order (setup_inputs dict order, params flattened in insertion order):
//  0 input_x | 1 conv1_w 2 conv1_b | 3..34 (conv2..conv17 w,b pairs per
//  CONV3_NAMES) | 35..38 ca1_{w1,b1,w2,b2} | 39..42 ca2 | 43 conv7_w 44 conv7_b
//  45 conv13_w 46 conv13_b | 47 off_w 48 off_b | 49 bn_g 50 bn_b 51 bn_m 52 bn_v
// ---------------------------------------------------------------------------
extern "C" void kernel_launch(void* const* d_in, const int* in_sizes, int n_in,
                              void* d_out, int out_size, void* d_ws, size_t ws_size,
                              hipStream_t stream) {
  (void)in_sizes; (void)n_in; (void)out_size; (void)ws_size;
  const float* x = (const float*)d_in[0];
  char* ws = (char*)d_ws;
  const size_t S = (size_t)BATCH * HH * WW * 64 * sizeof(float);  // 33.55 MB
  float* F[6];
  for (int i = 0; i < 6; ++i) F[i] = (float*)(ws + (size_t)i * S);
  size_t o = 6 * S;
  _Float16* WP3 = (_Float16*)(ws + o); o += (size_t)16 * 36864 * sizeof(_Float16);
  _Float16* WP1 = (_Float16*)(ws + o); o += (size_t)32 * 16384 * sizeof(_Float16);
  float* MEANS = (float*)(ws + o); o += 8192 * sizeof(float);
  float* SCALE = (float*)(ws + o); o += 8192 * sizeof(float);
  float* OFFB  = (float*)(ws + o); o += (size_t)BATCH * HH * WW * 18 * sizeof(float);

  dim3 cg(WW / 16, HH / 8, BATCH);  // (4, 8, 32)

  // Pack all sixteen 3x3 64->64 weight tensors into WMMA fragment layout.
  for (int l = 0; l < 16; ++l)
    pack_w3<<<144, 256, 0, stream>>>((const float*)d_in[3 + 2 * l],
                                     WP3 + (size_t)l * 36864);

  conv1_k<<<512, 256, 0, stream>>>(x, (const float*)d_in[1], (const float*)d_in[2], F[0]);

  auto C3 = [&](int slot, const float* in, float* out, const float* res,
                int relu, int add) {
    conv3_wmma<<<cg, 256, 0, stream>>>(in, WP3 + (size_t)slot * 36864,
                                       (const float*)d_in[4 + 2 * slot],
                                       res, out, relu, add);
  };

  C3(0,  F[0], F[1], F[0], 1, 0);  // conv2   = relu(conv(c1))
  C3(1,  F[1], F[2], F[0], 0, 1);  // conv3   = c1 + conv(c2)
  C3(2,  F[2], F[1], F[0], 1, 0);  // conv4
  C3(3,  F[1], F[3], F[2], 0, 1);  // conv5   = c3 + conv(c4)
  C3(4,  F[3], F[1], F[0], 1, 0);  // conv6
  C3(5,  F[1], F[4], F[3], 0, 1);  // conv6_1 + c5

  ca_mean<<<dim3(32, 4), 256, 0, stream>>>(F[4], F[3], F[2], F[0], MEANS);
  ca_scale<<<32, 256, 0, stream>>>(MEANS, (const float*)d_in[35], (const float*)d_in[36],
                                   (const float*)d_in[37], (const float*)d_in[38], SCALE);
  pack_w1<<<2048, 256, 0, stream>>>((const float*)d_in[43], SCALE, WP1);
  conv1x1_wmma<<<cg, 256, 0, stream>>>(F[4], F[3], F[2], F[0], WP1,
                                       (const float*)d_in[44], F[0], F[5]);  // conv7

  C3(6,  F[5], F[1], F[0], 1, 0);  // conv8
  C3(7,  F[1], F[0], F[5], 0, 1);  // conv9   = conv(c8) + c7
  C3(8,  F[0], F[1], F[0], 1, 0);  // conv10
  C3(9,  F[1], F[2], F[0], 0, 1);  // conv11  = conv(c10) + c9
  C3(10, F[2], F[1], F[0], 1, 0);  // conv12
  C3(11, F[1], F[3], F[2], 0, 1);  // conv12_1 + c11

  ca_mean<<<dim3(32, 4), 256, 0, stream>>>(F[3], F[2], F[0], F[5], MEANS);
  ca_scale<<<32, 256, 0, stream>>>(MEANS, (const float*)d_in[39], (const float*)d_in[40],
                                   (const float*)d_in[41], (const float*)d_in[42], SCALE);
  pack_w1<<<2048, 256, 0, stream>>>((const float*)d_in[45], SCALE, WP1);
  conv1x1_wmma<<<cg, 256, 0, stream>>>(F[3], F[2], F[0], F[5], WP1,
                                       (const float*)d_in[46], F[5], F[4]);  // conv13

  C3(12, F[4], F[1], F[0], 1, 0);  // conv14
  C3(13, F[1], F[0], F[4], 0, 1);  // conv15 = conv(c14) + c13
  C3(14, F[0], F[1], F[0], 1, 0);  // conv16
  C3(15, F[1], F[2], F[0], 0, 1);  // conv17 = conv(c16) + c15  -> feat

  off_conv<<<9216, 256, 0, stream>>>(F[2], (const float*)d_in[47], (const float*)d_in[48],
                                     (const float*)d_in[49], (const float*)d_in[50],
                                     (const float*)d_in[51], (const float*)d_in[52], OFFB);
  deform_k<<<73728, 256, 0, stream>>>(F[2], OFFB, (float*)d_out);
}
